// MultiStepEmpowerment_43224550867250
// MI455X (gfx1250) — matrix operand order repaired and checked by reading, not compiled, not run on hardware
//
#include <hip/hip_runtime.h>
#include <hip/hip_bf16.h>

// ---------------------------------------------------------------------------
// MI455X (gfx1250, wave32) implementation.
// Pipeline:
//   k_prep      : cast/pack weights to f16 (W_se, W_p1[:, :256], W_p2, [W_mu;W_lv])
//   k_gru       : 50 tiny GRU sequences -> P_a[64][256] = action_enc@W_p1a.T + b_p1
//   k_state_enc : WMMA  state_enc_h = f16(state@W_se.T + b_se)        [1024x256]
//   k_ps        : WMMA  P_s = state_enc_h @ W_p1s.T                   [1024x256] f32
//   k_main      : per-b fused: LN/relu -> WMMA GEMM (W_p2) -> WMMA GEMM (W_mu|W_lv)
//                 -> reparam sample -> WMMA Gram -> KNN entropy -> out[b]
// ---------------------------------------------------------------------------

typedef __attribute__((ext_vector_type(16))) _Float16 v16h;
typedef __attribute__((ext_vector_type(8)))  float    v8f;

#define NA_  50
#define HD_  256
#define SD_  512
#define LD_  64

union FragU { v16h v; float4 q[2]; };

// A fragment (16x32 f16, M x K): row = lane%16, k chunks at (lane/16)*8 and +16
__device__ __forceinline__ v16h loadA(const _Float16* base, int stride, int lane) {
  const _Float16* p = base + (lane & 15) * stride + ((lane >> 4) << 3);
  FragU u;
  u.q[0] = *(const float4*)(p);
  u.q[1] = *(const float4*)(p + 16);
  return u.v;
}

// B fragment (32x16 f16, K x N) from row-major W[n][k]: row n = lane%16,
// k chunks at (lane/16)*8 and +16
__device__ __forceinline__ v16h loadB(const _Float16* base, int stride, int lane) {
  const _Float16* p = base + (lane & 15) * stride + ((lane >> 4) << 3);
  FragU u;
  u.q[0] = *(const float4*)(p);
  u.q[1] = *(const float4*)(p + 16);
  return u.v;
}

// A fragment with on-the-fly f32 -> f16 conversion
__device__ __forceinline__ v16h loadA_f32(const float* base, int stride, int lane) {
  const float* p = base + (lane & 15) * stride + ((lane >> 4) << 3);
  v16h f;
#pragma unroll
  for (int i = 0; i < 8; ++i) {
    f[i]     = (_Float16)p[i];
    f[i + 8] = (_Float16)p[16 + i];
  }
  return f;
}

__device__ __forceinline__ v8f wmma_f16(v16h a, v16h b, v8f c) {
  return __builtin_amdgcn_wmma_f32_16x16x32_f16(false, a, false, b, (short)0, c,
                                                false, false);
}

// ---------------------------------------------------------------------------
// Weight prep: f32 -> f16 packs
// ---------------------------------------------------------------------------
__global__ void k_prep(const float* __restrict__ Wse, const float* __restrict__ Wp1,
                       const float* __restrict__ Wp2, const float* __restrict__ Wmu,
                       const float* __restrict__ Wlv,
                       _Float16* __restrict__ WseH, _Float16* __restrict__ Wp1sH,
                       _Float16* __restrict__ Wp2H, _Float16* __restrict__ WmlH) {
  int i = blockIdx.x * blockDim.x + threadIdx.x;  // 0 .. 524287
  if (i < 131072) {
    WseH[i] = (_Float16)Wse[i];                       // [256][512]
  } else if (i < 196608) {
    int j = i - 131072; int n = j >> 8, k = j & 255;  // W_p1[:, :256]
    Wp1sH[j] = (_Float16)Wp1[n * 320 + k];
  } else if (i < 262144) {
    int j = i - 196608;
    Wp2H[j] = (_Float16)Wp2[j];                       // [256][256]
  } else {
    int j = i - 262144; int n = j >> 8, k = j & 255;  // [W_mu; W_lv] -> [1024][256]
    WmlH[j] = (_Float16)(n < 512 ? Wmu[n * 256 + k] : Wlv[(n - 512) * 256 + k]);
  }
}

// ---------------------------------------------------------------------------
// GRU over 50 unique action sequences -> P_a[64][256] (rows >=50 zeroed)
// one block per sequence, 256 threads (thread t = hidden unit t)
// ---------------------------------------------------------------------------
__global__ void k_gru(const int* __restrict__ action_idx,
                      const float* __restrict__ g0Wih, const float* __restrict__ g0Whh,
                      const float* __restrict__ g0bih, const float* __restrict__ g0bhh,
                      const float* __restrict__ g1Wih, const float* __restrict__ g1Whh,
                      const float* __restrict__ g1bih, const float* __restrict__ g1bhh,
                      const float* __restrict__ Wae, const float* __restrict__ bae,
                      const float* __restrict__ Wp1, const float* __restrict__ bp1,
                      float* __restrict__ Pa) {
  __shared__ float h[HD_];
  __shared__ float y0[5][HD_];
  __shared__ float ae[LD_];
  int t = threadIdx.x;
  int na = blockIdx.x;
  if (na >= NA_) { Pa[na * HD_ + t] = 0.f; return; }

  h[t] = 0.f;
  __syncthreads();
  // layer 0: input = one-hot(action) -> column select of Wih
  for (int step = 0; step < 5; ++step) {
    int a = action_idx[na * 5 + step];
    float gir = g0Wih[t * 8 + a] + g0bih[t];
    float giz = g0Wih[(256 + t) * 8 + a] + g0bih[256 + t];
    float gin = g0Wih[(512 + t) * 8 + a] + g0bih[512 + t];
    float ghr = g0bhh[t], ghz = g0bhh[256 + t], ghn = g0bhh[512 + t];
    for (int k = 0; k < HD_; ++k) {
      float hk = h[k];
      ghr += g0Whh[t * 256 + k] * hk;
      ghz += g0Whh[(256 + t) * 256 + k] * hk;
      ghn += g0Whh[(512 + t) * 256 + k] * hk;
    }
    float r  = 1.f / (1.f + __expf(-(gir + ghr)));
    float z  = 1.f / (1.f + __expf(-(giz + ghz)));
    float n  = tanhf(gin + r * ghn);
    float hn = (1.f - z) * n + z * h[t];
    __syncthreads();
    h[t] = hn; y0[step][t] = hn;
    __syncthreads();
  }
  h[t] = 0.f;
  __syncthreads();
  // layer 1
  for (int step = 0; step < 5; ++step) {
    float gir = g1bih[t], giz = g1bih[256 + t], gin = g1bih[512 + t];
    float ghr = g1bhh[t], ghz = g1bhh[256 + t], ghn = g1bhh[512 + t];
    for (int k = 0; k < HD_; ++k) {
      float xk = y0[step][k], hk = h[k];
      gir += g1Wih[t * 256 + k] * xk;         ghr += g1Whh[t * 256 + k] * hk;
      giz += g1Wih[(256 + t) * 256 + k] * xk; ghz += g1Whh[(256 + t) * 256 + k] * hk;
      gin += g1Wih[(512 + t) * 256 + k] * xk; ghn += g1Whh[(512 + t) * 256 + k] * hk;
    }
    float r  = 1.f / (1.f + __expf(-(gir + ghr)));
    float z  = 1.f / (1.f + __expf(-(giz + ghz)));
    float n  = tanhf(gin + r * ghn);
    float hn = (1.f - z) * n + z * h[t];
    __syncthreads();
    h[t] = hn;
    __syncthreads();
  }
  if (t < LD_) {
    float s = bae[t];
    for (int k = 0; k < HD_; ++k) s += Wae[t * 256 + k] * h[k];
    ae[t] = s;
  }
  __syncthreads();
  // P_a row = action_enc @ W_p1[:, 256:].T + b_p1
  float s = bp1[t];
  for (int l = 0; l < LD_; ++l) s += ae[l] * Wp1[t * 320 + 256 + l];
  Pa[na * HD_ + t] = s;
}

// ---------------------------------------------------------------------------
// state_enc_h = f16(state @ W_se.T + b_se)   [1024 x 256], one wave per tile
// ---------------------------------------------------------------------------
__global__ void k_state_enc(const float* __restrict__ state,
                            const _Float16* __restrict__ WseH,
                            const float* __restrict__ b_se,
                            _Float16* __restrict__ SEh) {
  int lane = threadIdx.x & 31;
  int w = (blockIdx.x * blockDim.x + threadIdx.x) >> 5;  // 0..1023
  int mt = w >> 4, nt = w & 15;
  v8f acc = {};
  const float*    Ab = state + mt * 16 * SD_;
  const _Float16* Bb = WseH + nt * 16 * SD_;
  for (int kt = 0; kt < 16; ++kt) {
    v16h a = loadA_f32(Ab + kt * 32, SD_, lane);
    v16h b = loadB(Bb + kt * 32, SD_, lane);
    acc = wmma_f16(a, b, acc);
  }
  int col = nt * 16 + (lane & 15);
  int rb  = mt * 16 + ((lane >> 4) << 3);
#pragma unroll
  for (int v = 0; v < 8; ++v)
    SEh[(rb + v) * HD_ + col] = (_Float16)(acc[v] + b_se[col]);
}

// ---------------------------------------------------------------------------
// P_s = state_enc_h @ W_p1s.T   [1024 x 256] f32
// ---------------------------------------------------------------------------
__global__ void k_ps(const _Float16* __restrict__ SEh,
                     const _Float16* __restrict__ Wp1sH,
                     float* __restrict__ Ps) {
  int lane = threadIdx.x & 31;
  int w = (blockIdx.x * blockDim.x + threadIdx.x) >> 5;
  int mt = w >> 4, nt = w & 15;
  v8f acc = {};
  for (int kt = 0; kt < 8; ++kt) {
    v16h a = loadA(SEh + mt * 16 * HD_ + kt * 32, HD_, lane);
    v16h b = loadB(Wp1sH + nt * 16 * HD_ + kt * 32, HD_, lane);
    acc = wmma_f16(a, b, acc);
  }
  int col = nt * 16 + (lane & 15);
  int rb  = mt * 16 + ((lane >> 4) << 3);
#pragma unroll
  for (int v = 0; v < 8; ++v) Ps[(rb + v) * HD_ + col] = acc[v];
}

// ---------------------------------------------------------------------------
// Main fused kernel: one block per state b (1024 blocks, 256 threads = 8 waves)
// ---------------------------------------------------------------------------
__global__ void k_main(const _Float16* __restrict__ Wp2H,
                       const _Float16* __restrict__ WmlH,
                       const float* __restrict__ Ps, const float* __restrict__ Pa,
                       const float* __restrict__ ln_g, const float* __restrict__ ln_b,
                       const float* __restrict__ b_p2, const float* __restrict__ b_mu,
                       const float* __restrict__ b_lv, const float* __restrict__ eps,
                       float* __restrict__ out) {
  extern __shared__ char smem[];
  _Float16* H2 = (_Float16*)(smem);          // 64*256 f16 = 32768 B
  _Float16* H1 = (_Float16*)(smem + 32768);  // 64*256 f16 (dead after GEMM1)
  _Float16* X  = (_Float16*)(smem + 32768);  // 64*128 f16 (overlays H1)
  float* G     = (float*)(smem + 49152);     // 64*64 f32
  float* sqn   = (float*)(smem + 65536);     // 64 f32
  float* psrow = (float*)(smem + 65792);     // 256 f32
  float* red   = (float*)(smem + 66816);     // 256 f32

  int t = threadIdx.x, lane = t & 31, w = t >> 5;
  int b = blockIdx.x;
  psrow[t] = Ps[b * HD_ + t];
  if (t < 64) sqn[t] = 0.f;
  __syncthreads();

  // Phase A: H1 = relu(LayerNorm(P_s[b] + P_a[na]))  (rows >= 50 zero)
  for (int na = w; na < 64; na += 8) {
    if (na < NA_) {
      float vals[8]; float s = 0.f, s2 = 0.f;
#pragma unroll
      for (int u = 0; u < 8; ++u) {
        int j = lane * 8 + u;
        float v = psrow[j] + Pa[na * HD_ + j];
        vals[u] = v; s += v; s2 += v * v;
      }
#pragma unroll
      for (int off = 16; off > 0; off >>= 1) {
        s  += __shfl_xor(s, off, 32);
        s2 += __shfl_xor(s2, off, 32);
      }
      float mean = s * (1.f / 256.f);
      float var  = s2 * (1.f / 256.f) - mean * mean;
      float rstd = rsqrtf(var + 1e-5f);
#pragma unroll
      for (int u = 0; u < 8; ++u) {
        int j = lane * 8 + u;
        float hv = (vals[u] - mean) * rstd * ln_g[j] + ln_b[j];
        H1[na * HD_ + j] = (_Float16)fmaxf(hv, 0.f);
      }
    } else {
#pragma unroll
      for (int u = 0; u < 8; ++u) H1[na * HD_ + lane * 8 + u] = (_Float16)0.f;
    }
  }
  __syncthreads();

  // Phase B: H2 = relu(H1 @ W_p2.T + b_p2)   [64 x 256]
#pragma unroll
  for (int i = 0; i < 8; ++i) {
    int id = w * 8 + i, mt = id >> 4, nt = id & 15;
    v8f acc = {};
    for (int kt = 0; kt < 8; ++kt) {
      v16h a = loadA(H1 + mt * 16 * HD_ + kt * 32, HD_, lane);
      v16h bb = loadB(Wp2H + nt * 16 * HD_ + kt * 32, HD_, lane);
      acc = wmma_f16(a, bb, acc);
    }
    int col = nt * 16 + (lane & 15);
    int rb  = mt * 16 + ((lane >> 4) << 3);
#pragma unroll
    for (int v = 0; v < 8; ++v)
      H2[(rb + v) * HD_ + col] = (_Float16)fmaxf(acc[v] + b_p2[col], 0.f);
  }
  __syncthreads();

  // Phase C: reparameterized samples in 128-col chunks + Gram accumulation
  v8f g0 = {}, g1 = {};  // this wave's 2 Gram tiles, live across chunks
  int gid0 = w * 2, gid1 = w * 2 + 1;
  int gmt0 = gid0 >> 2, gnt0 = gid0 & 3, gmt1 = gid1 >> 2, gnt1 = gid1 & 3;

  for (int cc = 0; cc < 4; ++cc) {
#pragma unroll
    for (int i = 0; i < 4; ++i) {  // 32 (mu,lv) tile pairs / 8 waves
      int id = w * 4 + i, mt = id >> 3, ct = id & 7;
      int ntmu = cc * 8 + ct;  // mu column-tile in [0,32)
      v8f am = {}, al = {};
      for (int kt = 0; kt < 8; ++kt) {
        v16h a  = loadA(H2 + mt * 16 * HD_ + kt * 32, HD_, lane);
        v16h bm = loadB(WmlH + ntmu * 16 * HD_ + kt * 32, HD_, lane);
        v16h bl = loadB(WmlH + (512 + ntmu * 16) * HD_ + kt * 32, HD_, lane);
        am = wmma_f16(a, bm, am);
        al = wmma_f16(a, bl, al);
      }
      int colg = ntmu * 16 + (lane & 15);  // 0..511
      int colx = colg - cc * 128;          // 0..127
      int rb   = mt * 16 + ((lane >> 4) << 3);
#pragma unroll
      for (int v = 0; v < 8; ++v) {
        int row = rb + v;
        float x = 0.f;
        if (row < NA_) {
          float mu = am[v] + b_mu[colg];
          float lv = al[v] + b_lv[colg];
          float e  = eps[((size_t)(b * NA_ + row)) * SD_ + colg];
          x = mu + e * __expf(0.5f * lv);
          atomicAdd(&sqn[row], x * x);
        }
        X[row * 128 + colx] = (_Float16)x;
      }
    }
    __syncthreads();
    // Gram += X_chunk @ X_chunk.T (B frag loads identically from row-major X)
    for (int kt = 0; kt < 4; ++kt) {
      v16h a0 = loadA(X + gmt0 * 16 * 128 + kt * 32, 128, lane);
      v16h b0 = loadB(X + gnt0 * 16 * 128 + kt * 32, 128, lane);
      g0 = wmma_f16(a0, b0, g0);
      v16h a1 = loadA(X + gmt1 * 16 * 128 + kt * 32, 128, lane);
      v16h b1 = loadB(X + gnt1 * 16 * 128 + kt * 32, 128, lane);
      g1 = wmma_f16(a1, b1, g1);
    }
    __syncthreads();
  }
  // dump Gram tiles
  {
    int col = gnt0 * 16 + (lane & 15), rb = gmt0 * 16 + ((lane >> 4) << 3);
#pragma unroll
    for (int v = 0; v < 8; ++v) G[(rb + v) * 64 + col] = g0[v];
    col = gnt1 * 16 + (lane & 15); rb = gmt1 * 16 + ((lane >> 4) << 3);
#pragma unroll
    for (int v = 0; v < 8; ++v) G[(rb + v) * 64 + col] = g1[v];
  }
  __syncthreads();

  // Phase D: KNN (k=5) Kozachenko-Leonenko entropy over 50 samples
  float lr = 0.f;
  if (t < NA_) {
    float k0 = 3e38f, k1 = 3e38f, k2 = 3e38f, k3 = 3e38f, k4 = 3e38f;
    float ni = sqn[t];
    for (int j = 0; j < NA_; ++j) {
      if (j == t) continue;
      float d2 = fmaxf(ni + sqn[j] - 2.f * G[t * 64 + j], 0.f);
      if (d2 < k4) {
        k4 = d2;
        if (k4 < k3) { float tm = k3; k3 = k4; k4 = tm;
          if (k3 < k2) { tm = k2; k2 = k3; k3 = tm;
            if (k2 < k1) { tm = k1; k1 = k2; k2 = tm;
              if (k1 < k0) { tm = k0; k0 = k1; k1 = tm; } } } }
      }
    }
    lr = 0.5f * logf(fmaxf(k4, 1e-12f));  // log of 5th-NN distance
  }
  red[t] = lr;
  __syncthreads();
  for (int s = 128; s > 0; s >>= 1) {
    if (t < s) red[t] += red[t + s];
    __syncthreads();
  }
  if (t == 0) {
    // digamma(50) - digamma(5) + 256*ln(pi) - lnGamma(257) = -871.8105556
    float ent = -871.8105556f + 512.f * (red[0] * (1.f / 50.f));
    out[b] = ent / sqrtf(1.f + 1e-8f);
  }
}

// ---------------------------------------------------------------------------
extern "C" void kernel_launch(void* const* d_in, const int* in_sizes, int n_in,
                              void* d_out, int out_size, void* d_ws, size_t ws_size,
                              hipStream_t stream) {
  const float* state      = (const float*)d_in[0];
  const int*   action_idx = (const int*)d_in[1];
  const float* eps        = (const float*)d_in[2];
  const float* W_se  = (const float*)d_in[3];
  const float* b_se  = (const float*)d_in[4];
  const float* g0Wih = (const float*)d_in[5];
  const float* g0Whh = (const float*)d_in[6];
  const float* g0bih = (const float*)d_in[7];
  const float* g0bhh = (const float*)d_in[8];
  const float* g1Wih = (const float*)d_in[9];
  const float* g1Whh = (const float*)d_in[10];
  const float* g1bih = (const float*)d_in[11];
  const float* g1bhh = (const float*)d_in[12];
  const float* W_ae  = (const float*)d_in[13];
  const float* b_ae  = (const float*)d_in[14];
  const float* W_p1  = (const float*)d_in[15];
  const float* b_p1  = (const float*)d_in[16];
  const float* ln_g  = (const float*)d_in[17];
  const float* ln_b  = (const float*)d_in[18];
  const float* W_p2  = (const float*)d_in[19];
  const float* b_p2  = (const float*)d_in[20];
  const float* W_mu  = (const float*)d_in[21];
  const float* b_mu  = (const float*)d_in[22];
  const float* W_lv  = (const float*)d_in[23];
  const float* b_lv  = (const float*)d_in[24];
  float* out = (float*)d_out;

  char* ws = (char*)d_ws;
  _Float16* WseH  = (_Float16*)(ws + 0);        // 256*512 f16
  _Float16* Wp1sH = (_Float16*)(ws + 262144);   // 256*256 f16
  _Float16* Wp2H  = (_Float16*)(ws + 393216);   // 256*256 f16
  _Float16* WmlH  = (_Float16*)(ws + 524288);   // 1024*256 f16
  _Float16* SEh   = (_Float16*)(ws + 1048576);  // 1024*256 f16
  float*    Ps    = (float*)   (ws + 1572864);  // 1024*256 f32
  float*    Pa    = (float*)   (ws + 2621440);  // 64*256 f32

  k_prep<<<2048, 256, 0, stream>>>(W_se, W_p1, W_p2, W_mu, W_lv,
                                   WseH, Wp1sH, Wp2H, WmlH);
  k_gru<<<64, 256, 0, stream>>>(action_idx, g0Wih, g0Whh, g0bih, g0bhh,
                                g1Wih, g1Whh, g1bih, g1bhh,
                                W_ae, b_ae, W_p1, b_p1, Pa);
  k_state_enc<<<128, 256, 0, stream>>>(state, WseH, b_se, SEh);
  k_ps<<<128, 256, 0, stream>>>(SEh, Wp1sH, Ps);
  k_main<<<1024, 256, 67840, stream>>>(Wp2H, WmlH, Ps, Pa, ln_g, ln_b,
                                       b_p2, b_mu, b_lv, eps, out);
}
